// RQVAE_76244259439100
// MI455X (gfx1250) — matrix operand order repaired and checked by reading, not compiled, not run on hardware
//
#include <hip/hip_runtime.h>
#include <cstdint>
#include <cstddef>

// ---------------------------------------------------------------------------
// CDNA5 (gfx1250) WMMA types
// ---------------------------------------------------------------------------
typedef __attribute__((ext_vector_type(16))) __bf16 v16bf;
typedef __attribute__((ext_vector_type(8)))  __bf16 v8bf;
typedef __attribute__((ext_vector_type(8)))  float  v8f;

__device__ __forceinline__ v8f wmma_bf16(v16bf a, v16bf b, v8f c) {
  // D(16x16,f32) = A(16x32,bf16) * B(32x16,bf16) + C
  return __builtin_amdgcn_wmma_f32_16x16x32_bf16(false, a, false, b, (short)0, c,
                                                 false, false);
}

#define ACT_NONE  0
#define ACT_RELU  1
#define ACT_LRELU 2

// ---------------------------------------------------------------------------
// Implicit-GEMM conv. One wave computes MT vertical 16x16 tiles: a
// (16*MT Cout) x (16 pixel) strip, so the activation (B) fragments are loaded
// ONCE per K-chunk and reused by MT WMMAs (weights are tiny / L2-resident).
//
// Activations: channel-last bf16 [B][H][W][C].  Weights repacked bf16
// [CoutPad][K] with K = (ky*KW+kx)*Cin + ci  (tap-major, channel-minor).
// A fragment (16x32 bf16): lane l holds M=l&15; halfs 0..7 = K c+16h..c+16h+7,
// halfs 8..15 = K c+16h+8..c+16h+15  (h = l>=16).
// B fragment (32x16 bf16): lane l holds N=l&15; halfs 0..15 = K c+16h..+15.
// C/D (16x16 f32): lane l, elem j -> M = j + 8h, N = l&15.
// ---------------------------------------------------------------------------
struct ConvParams {
  const __bf16* in;
  const __bf16* w;
  const float*  bias;     // may be null
  const __bf16* resid;    // may be null (residual add, same layout as out)
  __bf16*       out;      // channel-last bf16, may be null
  float*        out_f32;  // NCHW f32 (final recons), may be null
  int B, IH, IW, Cin, Cout, OH, OW;
  int KH, KW, stride, padY, padX;
  int OHfull, OWfull, oyMul, oyOff, oxMul, oxOff;
  int act;
  int Ktot;
};

template <int MT, bool VEC32>   // VEC32: Cin % 32 == 0 -> vectorized tap path
__global__ void conv_wmma(ConvParams p) {
  const int lane = threadIdx.x & 31;
  const int hf   = lane >> 4;     // 0/1 half of wave
  const int l15  = lane & 15;

  const int n   = blockIdx.x * 16 + l15;   // pixel (GEMM column)
  const int ohw = p.OH * p.OW;
  const int bN  = n / ohw;
  const int rem = n - bN * ohw;
  const int oy  = rem / p.OW;
  const int ox  = rem - oy * p.OW;
  const int iy0 = oy * p.stride - p.padY;
  const int ix0 = ox * p.stride - p.padX;

  const int  mBase    = blockIdx.y * (16 * MT);
  const long wRowBase = (long)(mBase + l15) * p.Ktot;
  const long wTileStp = 16L * p.Ktot;

  v8bf bz;
#pragma unroll
  for (int t = 0; t < 8; ++t) bz[t] = (__bf16)0.0f;

  v8f accs[MT];
#pragma unroll
  for (int t = 0; t < MT; ++t)
#pragma unroll
    for (int j = 0; j < 8; ++j) accs[t][j] = 0.0f;

  if (VEC32) {
    int tap = 0;
    for (int ky = 0; ky < p.KH; ++ky) {
      const int  iy  = iy0 + ky;
      const bool yok = (iy >= 0) & (iy < p.IH);
      for (int kx = 0; kx < p.KW; ++kx, ++tap) {
        const int  ix = ix0 + kx;
        const bool ok = yok & (ix >= 0) & (ix < p.IW);
        const __bf16* src = p.in + ((long)(bN * p.IH + iy) * p.IW + ix) * p.Cin;
        if (ok) __builtin_prefetch((const void*)(src + p.Cin), 0, 0);  // next tap (x+1)
        for (int cc = 0; cc < p.Cin; cc += 32) {
          const int c = tap * p.Cin + cc;
          v8bf b0 = bz, b1 = bz;
          if (ok) {
            b0 = *(const v8bf*)(src + cc + hf * 16);
            b1 = *(const v8bf*)(src + cc + hf * 16 + 8);
          }
          v16bf bf = __builtin_shufflevector(b0, b1, 0,1,2,3,4,5,6,7,8,9,10,11,12,13,14,15);
#pragma unroll
          for (int t = 0; t < MT; ++t) {
            const __bf16* wp = p.w + wRowBase + (long)t * wTileStp + c;
            v8bf a0 = *(const v8bf*)(wp + hf * 16);
            v8bf a1 = *(const v8bf*)(wp + hf * 16 + 8);
            v16bf af = __builtin_shufflevector(a0, a1, 0,1,2,3,4,5,6,7,8,9,10,11,12,13,14,15);
            accs[t] = wmma_bf16(af, bf, accs[t]);
          }
        }
      }
    }
  } else {
    for (int c = 0; c < p.Ktot; c += 32) {
      v16bf bf;
#pragma unroll
      for (int t = 0; t < 16; ++t) {
        const int k   = c + hf * 16 + t;
        const int tap = k / p.Cin;
        const int ci  = k - tap * p.Cin;
        const int ky  = tap / p.KW;
        const int kx  = tap - ky * p.KW;
        const int iy  = iy0 + ky, ix = ix0 + kx;
        __bf16 v = (__bf16)0.0f;
        if (iy >= 0 && iy < p.IH && ix >= 0 && ix < p.IW)
          v = p.in[((long)(bN * p.IH + iy) * p.IW + ix) * p.Cin + ci];
        bf[t] = v;
      }
#pragma unroll
      for (int t = 0; t < MT; ++t) {
        const __bf16* wp = p.w + wRowBase + (long)t * wTileStp + c;
        v8bf a0 = *(const v8bf*)(wp + hf * 16);
        v8bf a1 = *(const v8bf*)(wp + hf * 16 + 8);
        v16bf af = __builtin_shufflevector(a0, a1, 0,1,2,3,4,5,6,7,8,9,10,11,12,13,14,15);
        accs[t] = wmma_bf16(af, bf, accs[t]);
      }
    }
  }

  // -------- epilogue: bias, residual, activation, stores --------
  const int  mOff = hf * 8;
  const int  yy   = oy * p.oyMul + p.oyOff;
  const int  xx   = ox * p.oxMul + p.oxOff;
  const long pix  = (long)(bN * p.OHfull + yy) * p.OWfull + xx;

#pragma unroll
  for (int t = 0; t < MT; ++t) {
    const int mT = mBase + t * 16 + mOff;
    float vals[8];
#pragma unroll
    for (int j = 0; j < 8; ++j) {
      const int m = mT + j;
      float v = accs[t][j];
      if (p.bias && m < p.Cout) v += p.bias[m];
      vals[j] = v;
    }
    if (p.resid) {
      v8bf rv = *(const v8bf*)(p.resid + pix * p.Cout + mT);
#pragma unroll
      for (int j = 0; j < 8; ++j) vals[j] += (float)rv[j];
    }
#pragma unroll
    for (int j = 0; j < 8; ++j) {
      float v = vals[j];
      if (p.act == ACT_RELU)  v = v > 0.0f ? v : 0.0f;
      if (p.act == ACT_LRELU) v = v > 0.0f ? v : 0.01f * v;
      vals[j] = v;
    }
    if (p.out) {
      v8bf ov;
#pragma unroll
      for (int j = 0; j < 8; ++j) ov[j] = (__bf16)vals[j];
      *(v8bf*)(p.out + pix * p.Cout + mT) = ov;
    }
    if (p.out_f32) {
#pragma unroll
      for (int j = 0; j < 8; ++j) {
        const int m = mT + j;
        if (m < p.Cout)
          p.out_f32[((long)(bN * p.Cout + m) * p.OHfull + yy) * p.OWfull + xx] = vals[j];
      }
    }
  }
}

// ---------------------------------------------------------------------------
// Residual VQ: each wave owns 16 tokens held in the WMMA A-fragment layout.
// Per stage: 32 code tiles x (2 chained WMMAs, K=64) -> distances -> argmin.
// ---------------------------------------------------------------------------
#define VQ_WAVES 8
__global__ void vq_kernel(const __bf16* __restrict__ z, const float* __restrict__ cb,
                          const __bf16* __restrict__ cb16, const float* __restrict__ cbn,
                          __bf16* __restrict__ qout, float* __restrict__ outIdx,
                          float* __restrict__ outLoss, int Ntok) {
  __shared__ float sNorm[VQ_WAVES][16];
  __shared__ int   sIdx [VQ_WAVES][16];

  const int lane = threadIdx.x & 31;
  const int wave = threadIdx.x >> 5;
  const int hf   = lane >> 4;
  const int l15  = lane & 15;
  const int tok  = blockIdx.x * (VQ_WAVES * 16) + wave * 16 + l15;

  float r[32], qa[32];
#pragma unroll
  for (int s = 0; s < 32; ++s) qa[s] = 0.0f;

  // load z (lane holds half the channels of token l15): slot s -> d = s + 8*((s>>3)+hf)
#pragma unroll
  for (int run = 0; run < 4; ++run) {
    const int d0 = run * 16 + hf * 8;
    v8bf v = *(const v8bf*)(z + (long)tok * 64 + d0);
#pragma unroll
    for (int t = 0; t < 8; ++t) r[run * 8 + t] = (float)v[t];
  }

  const float invND = 1.0f / ((float)Ntok * 64.0f);

  for (int st = 0; st < 4; ++st) {
    // per-token |r|^2 (combine the two channel-halves)
    float nrm = 0.0f;
#pragma unroll
    for (int s = 0; s < 32; ++s) nrm += r[s] * r[s];
    nrm += __shfl_xor(nrm, 16, 32);
    if (hf == 0) sNorm[wave][l15] = nrm;
    __syncthreads();
    float tn[8];
#pragma unroll
    for (int j = 0; j < 8; ++j) tn[j] = sNorm[wave][j + 8 * hf];

    // A fragments from residual
    v16bf a0f, a1f;
#pragma unroll
    for (int t = 0; t < 16; ++t) { a0f[t] = (__bf16)r[t]; a1f[t] = (__bf16)r[16 + t]; }

    float bestd[8]; int besti[8];
#pragma unroll
    for (int j = 0; j < 8; ++j) { bestd[j] = 3.0e38f; besti[j] = 0; }

    const __bf16* cbs  = cb16 + (long)st * 512 * 64;
    const float*  cbns = cbn + st * 512;

    for (int tile = 0; tile < 32; ++tile) {
      const int  code  = tile * 16 + l15;
      const long cbase = (long)code * 64;
      v8bf c00 = *(const v8bf*)(cbs + cbase + hf * 8);
      v8bf c01 = *(const v8bf*)(cbs + cbase + hf * 8 + 16);
      v8bf c10 = *(const v8bf*)(cbs + cbase + 32 + hf * 8);
      v8bf c11 = *(const v8bf*)(cbs + cbase + 32 + hf * 8 + 16);
      v16bf b0f = __builtin_shufflevector(c00, c01, 0,1,2,3,4,5,6,7,8,9,10,11,12,13,14,15);
      v16bf b1f = __builtin_shufflevector(c10, c11, 0,1,2,3,4,5,6,7,8,9,10,11,12,13,14,15);
      v8f dacc;
#pragma unroll
      for (int j = 0; j < 8; ++j) dacc[j] = 0.0f;
      dacc = wmma_bf16(a0f, b0f, dacc);
      dacc = wmma_bf16(a1f, b1f, dacc);
      const float cn = cbns[code];
#pragma unroll
      for (int j = 0; j < 8; ++j) {
        const float dist = tn[j] - 2.0f * dacc[j] + cn;
        if (dist < bestd[j]) { bestd[j] = dist; besti[j] = code; }
      }
    }
    // argmin across the 16 lanes of each half-wave
#pragma unroll
    for (int mask = 1; mask < 16; mask <<= 1) {
#pragma unroll
      for (int j = 0; j < 8; ++j) {
        float od = __shfl_xor(bestd[j], mask, 32);
        int   oi = __shfl_xor(besti[j], mask, 32);
        if (od < bestd[j] || (od == bestd[j] && oi < besti[j])) { bestd[j] = od; besti[j] = oi; }
      }
    }
    if (l15 == 0) {
#pragma unroll
      for (int j = 0; j < 8; ++j) sIdx[wave][j + 8 * hf] = besti[j];
    }
    __syncthreads();
    const int myIdx = sIdx[wave][l15];
    if (hf == 0) outIdx[(long)tok * 4 + st] = (float)myIdx;

    // gather q (f32 codebook), accumulate forward value / loss, update residual
    const float* qrow = cb + ((long)st * 512 + myIdx) * 64;
    float lossp = 0.0f;
#pragma unroll
    for (int s = 0; s < 32; ++s) {
      const int d  = s + 8 * ((s >> 3) + hf);
      const float qv   = qrow[d];
      const float diff = qv - r[s];
      lossp += diff * diff;
      qa[s] += qv;
      r[s]  -= qv;
    }
#pragma unroll
    for (int mask = 1; mask < 32; mask <<= 1) lossp += __shfl_xor(lossp, mask, 32);
    if (lane == 0) atomicAdd(outLoss + st, 0.25f * lossp * invND);
    __syncthreads();
  }

  // write q_out (decoder input), channel-last bf16
#pragma unroll
  for (int run = 0; run < 4; ++run) {
    const int d0 = run * 16 + hf * 8;
    v8bf v;
#pragma unroll
    for (int t = 0; t < 8; ++t) v[t] = (__bf16)qa[run * 8 + t];
    *(v8bf*)(qout + (long)tok * 64 + d0) = v;
  }
}

// ---------------------------------------------------------------------------
// Small prep kernels
// ---------------------------------------------------------------------------
__global__ void repack_w(const float* __restrict__ w, __bf16* __restrict__ out,
                         int Cout, int CoutP, int Cin, int KH, int KW) {
  const long K = (long)KH * KW * Cin;
  const long t = (long)blockIdx.x * blockDim.x + threadIdx.x;
  if (t >= (long)CoutP * K) return;
  const int co = (int)(t / K);
  const int k  = (int)(t - (long)co * K);
  float v = 0.0f;
  if (co < Cout) {
    const int tap = k / Cin, ci = k - tap * Cin;
    const int ky = tap / KW, kx = tap - ky * KW;
    v = w[(((long)co * Cin + ci) * KH + ky) * KW + kx];
  }
  out[t] = (__bf16)v;
}

// convT(k4,s2,p1) parity sub-weights; w layout (Cin, Cout, 4, 4)
__global__ void repack_wT(const float* __restrict__ w, __bf16* __restrict__ out,
                          int Cin, int Cout, int CoutP) {
  const long K = 4L * Cin;
  const long t = (long)blockIdx.x * blockDim.x + threadIdx.x;
  if (t >= 4L * CoutP * K) return;
  const int par = (int)(t / (CoutP * K));
  long rem = t - (long)par * CoutP * K;
  const int co = (int)(rem / K);
  const int k  = (int)(rem - (long)co * K);
  const int oyp = par >> 1, oxp = par & 1;
  float v = 0.0f;
  if (co < Cout) {
    const int tap = k / Cin, ci = k - tap * Cin;
    const int ty = tap >> 1, tx = tap & 1;
    const int ky = (oyp == 0 ? 3 : 2) - 2 * ty;
    const int kx = (oxp == 0 ? 3 : 2) - 2 * tx;
    v = w[(((long)ci * Cout + co) * 4 + ky) * 4 + kx];
  }
  out[t] = (__bf16)v;
}

__global__ void cb_prep(const float* __restrict__ cb, __bf16* __restrict__ cb16,
                        float* __restrict__ cbn) {
  const int i = blockIdx.x * blockDim.x + threadIdx.x;
  if (i >= 4 * 512) return;
  const float* row = cb + (long)i * 64;
  float s = 0.0f;
  for (int d = 0; d < 64; ++d) {
    const float v = row[d];
    s += v * v;
    cb16[(long)i * 64 + d] = (__bf16)v;
  }
  cbn[i] = s;
}

__global__ void to_nhwc(const float* __restrict__ in, __bf16* __restrict__ out,
                        int B, int C, int H, int W) {
  const long t = (long)blockIdx.x * blockDim.x + threadIdx.x;
  if (t >= (long)B * C * H * W) return;
  const int c = (int)(t % C);
  long r = t / C;
  const int x = (int)(r % W); r /= W;
  const int y = (int)(r % H);
  const int b = (int)(r / H);
  out[t] = (__bf16)in[(((long)b * C + c) * H + y) * W + x];
}

__global__ void lrelu_bf16(__bf16* buf, long n) {
  const long t = (long)blockIdx.x * blockDim.x + threadIdx.x;
  if (t < n) {
    const float v = (float)buf[t];
    buf[t] = (__bf16)(v > 0.0f ? v : 0.01f * v);
  }
}

__global__ void zero_loss(float* p) {
  if (threadIdx.x < 4) p[threadIdx.x] = 0.0f;
}

// ---------------------------------------------------------------------------
// Host-side launcher
// ---------------------------------------------------------------------------
static void run_conv(const __bf16* in, const __bf16* w, const float* bias,
                     const __bf16* resid, __bf16* out, float* outf,
                     int B, int IH, int IW, int Cin, int Cout, int CoutP,
                     int OH, int OW, int KH, int KW, int stride, int padY, int padX,
                     int OHf, int OWf, int oyMul, int oyOff, int oxMul, int oxOff,
                     int act, hipStream_t s) {
  ConvParams p;
  p.in = in; p.w = w; p.bias = bias; p.resid = resid; p.out = out; p.out_f32 = outf;
  p.B = B; p.IH = IH; p.IW = IW; p.Cin = Cin; p.Cout = Cout; p.OH = OH; p.OW = OW;
  p.KH = KH; p.KW = KW; p.stride = stride; p.padY = padY; p.padX = padX;
  p.OHfull = OHf; p.OWfull = OWf; p.oyMul = oyMul; p.oyOff = oyOff;
  p.oxMul = oxMul; p.oxOff = oxOff; p.act = act; p.Ktot = KH * KW * Cin;

  const int mtot = CoutP / 16;   // all M-tiles handled by one wave
  dim3 grid((unsigned)((B * OH * OW) / 16), 1);
  if (Cin % 32 == 0) {
    if (mtot == 8)      conv_wmma<8, true><<<grid, 32, 0, s>>>(p);
    else if (mtot == 4) conv_wmma<4, true><<<grid, 32, 0, s>>>(p);
    else                conv_wmma<1, true><<<grid, 32, 0, s>>>(p);
  } else {
    conv_wmma<4, false><<<grid, 32, 0, s>>>(p);   // e1 only (Cout=64)
  }
}

extern "C" void kernel_launch(void* const* d_in, const int* in_sizes, int n_in,
                              void* d_out, int out_size, void* d_ws, size_t ws_size,
                              hipStream_t stream) {
  const float* inputs = (const float*)d_in[0];
  const float* e1_w = (const float*)d_in[1];  const float* e1_b = (const float*)d_in[2];
  const float* e2_w = (const float*)d_in[3];  const float* e2_b = (const float*)d_in[4];
  const float* e3_w = (const float*)d_in[5];  const float* e3_b = (const float*)d_in[6];
  const float* r1a_w = (const float*)d_in[7]; const float* r1b_w = (const float*)d_in[8];
  const float* r2a_w = (const float*)d_in[9]; const float* r2b_w = (const float*)d_in[10];
  const float* e4_w = (const float*)d_in[11]; const float* e4_b = (const float*)d_in[12];
  const float* cbks = (const float*)d_in[13];
  const float* d1_w = (const float*)d_in[14]; const float* d1_b = (const float*)d_in[15];
  const float* dr1a_w = (const float*)d_in[16]; const float* dr1b_w = (const float*)d_in[17];
  const float* dr2a_w = (const float*)d_in[18]; const float* dr2b_w = (const float*)d_in[19];
  const float* dt1_w = (const float*)d_in[20]; const float* dt1_b = (const float*)d_in[21];
  const float* dt2_w = (const float*)d_in[22]; const float* dt2_b = (const float*)d_in[23];
  (void)in_sizes; (void)n_in; (void)out_size;

  float* out_f = (float*)d_out;
  const long reconsN = 16L * 4 * 256 * 256;       // 4,194,304
  const long idxOff  = reconsN;                    // 65,536 indices
  const long lossOff = reconsN + 65536;

  // ---- workspace layout ----
  char* base = (char*)d_ws;
  size_t off = 0;
  auto take = [&](size_t elems) -> __bf16* {
    __bf16* p = (__bf16*)(base + off);
    off += ((elems * 2 + 255) / 256) * 256;
    return p;
  };
  __bf16* XN   = take(4194304);          // input NHWC bf16 (16,256,256,4)
  __bf16* bufA = take(16777216);
  __bf16* bufB = take(16777216);
  __bf16* bufC = take(16777216);
  __bf16* wp_e1  = take(64   * 64);
  __bf16* wp_e2  = take(128  * 1024);
  __bf16* wp_e3  = take(128  * 1152);
  __bf16* wp_r1a = take(128  * 1152);
  __bf16* wp_r1b = take(128  * 128);
  __bf16* wp_r2a = take(128  * 1152);
  __bf16* wp_r2b = take(128  * 128);
  __bf16* wp_e4  = take(64   * 128);
  __bf16* wp_d1  = take(128  * 576);
  __bf16* wp_dr1a= take(128  * 1152);
  __bf16* wp_dr1b= take(128  * 128);
  __bf16* wp_dr2a= take(128  * 1152);
  __bf16* wp_dr2b= take(128  * 128);
  __bf16* wp_dt1 = take(4 * 64 * 512);
  __bf16* wp_dt2 = take(4 * 16 * 256);
  __bf16* cb16   = take(4 * 512 * 64);
  float*  cbn    = (float*)(base + off); off += ((2048 * 4 + 255) / 256) * 256;
  if (off > ws_size) return;  // workspace too small; bail deterministically

  // ---- prep ----
  to_nhwc<<<(4194304 + 255) / 256, 256, 0, stream>>>(inputs, XN, 16, 4, 256, 256);
  auto rpk = [&](const float* w, __bf16* o, int Co, int CoP, int Ci, int KH, int KW) {
    long tot = (long)CoP * KH * KW * Ci;
    repack_w<<<(unsigned)((tot + 255) / 256), 256, 0, stream>>>(w, o, Co, CoP, Ci, KH, KW);
  };
  rpk(e1_w,  wp_e1,  64, 64, 4, 4, 4);
  rpk(e2_w,  wp_e2,  128, 128, 64, 4, 4);
  rpk(e3_w,  wp_e3,  128, 128, 128, 3, 3);
  rpk(r1a_w, wp_r1a, 128, 128, 128, 3, 3);
  rpk(r1b_w, wp_r1b, 128, 128, 128, 1, 1);
  rpk(r2a_w, wp_r2a, 128, 128, 128, 3, 3);
  rpk(r2b_w, wp_r2b, 128, 128, 128, 1, 1);
  rpk(e4_w,  wp_e4,  64, 64, 128, 1, 1);
  rpk(d1_w,  wp_d1,  128, 128, 64, 3, 3);
  rpk(dr1a_w, wp_dr1a, 128, 128, 128, 3, 3);
  rpk(dr1b_w, wp_dr1b, 128, 128, 128, 1, 1);
  rpk(dr2a_w, wp_dr2a, 128, 128, 128, 3, 3);
  rpk(dr2b_w, wp_dr2b, 128, 128, 128, 1, 1);
  repack_wT<<<(4 * 64 * 512 + 255) / 256, 256, 0, stream>>>(dt1_w, wp_dt1, 128, 64, 64);
  repack_wT<<<(4 * 16 * 256 + 255) / 256, 256, 0, stream>>>(dt2_w, wp_dt2, 64, 4, 16);
  cb_prep<<<(2048 + 255) / 256, 256, 0, stream>>>(cbks, cb16, cbn);
  zero_loss<<<1, 32, 0, stream>>>(out_f + lossOff);

  // ---- encoder ----
  run_conv(XN, wp_e1, e1_b, nullptr, bufA, nullptr, 16, 256, 256, 4, 64, 64,
           128, 128, 4, 4, 2, 1, 1, 128, 128, 1, 0, 1, 0, ACT_LRELU, stream);
  run_conv(bufA, wp_e2, e2_b, nullptr, bufB, nullptr, 16, 128, 128, 64, 128, 128,
           64, 64, 4, 4, 2, 1, 1, 64, 64, 1, 0, 1, 0, ACT_LRELU, stream);
  run_conv(bufB, wp_e3, e3_b, nullptr, bufC, nullptr, 16, 64, 64, 128, 128, 128,
           64, 64, 3, 3, 1, 1, 1, 64, 64, 1, 0, 1, 0, ACT_LRELU, stream);
  // res1: x=C
  run_conv(bufC, wp_r1a, nullptr, nullptr, bufB, nullptr, 16, 64, 64, 128, 128, 128,
           64, 64, 3, 3, 1, 1, 1, 64, 64, 1, 0, 1, 0, ACT_RELU, stream);
  run_conv(bufB, wp_r1b, nullptr, bufC, bufA, nullptr, 16, 64, 64, 128, 128, 128,
           64, 64, 1, 1, 1, 0, 0, 64, 64, 1, 0, 1, 0, ACT_NONE, stream);
  // res2: x=A
  run_conv(bufA, wp_r2a, nullptr, nullptr, bufB, nullptr, 16, 64, 64, 128, 128, 128,
           64, 64, 3, 3, 1, 1, 1, 64, 64, 1, 0, 1, 0, ACT_RELU, stream);
  run_conv(bufB, wp_r2b, nullptr, bufA, bufC, nullptr, 16, 64, 64, 128, 128, 128,
           64, 64, 1, 1, 1, 0, 0, 64, 64, 1, 0, 1, 0, ACT_NONE, stream);
  lrelu_bf16<<<(8388608 + 255) / 256, 256, 0, stream>>>(bufC, 8388608L);
  run_conv(bufC, wp_e4, e4_b, nullptr, bufB, nullptr, 16, 64, 64, 128, 64, 64,
           64, 64, 1, 1, 1, 0, 0, 64, 64, 1, 0, 1, 0, ACT_LRELU, stream);

  // ---- residual VQ: z = bufB, q_out -> bufA ----
  vq_kernel<<<65536 / (VQ_WAVES * 16), VQ_WAVES * 32, 0, stream>>>(
      bufB, cbks, cb16, cbn, bufA, out_f + idxOff, out_f + lossOff, 65536);

  // ---- decoder ----
  run_conv(bufA, wp_d1, d1_b, nullptr, bufC, nullptr, 16, 64, 64, 64, 128, 128,
           64, 64, 3, 3, 1, 1, 1, 64, 64, 1, 0, 1, 0, ACT_LRELU, stream);
  run_conv(bufC, wp_dr1a, nullptr, nullptr, bufB, nullptr, 16, 64, 64, 128, 128, 128,
           64, 64, 3, 3, 1, 1, 1, 64, 64, 1, 0, 1, 0, ACT_RELU, stream);
  run_conv(bufB, wp_dr1b, nullptr, bufC, bufA, nullptr, 16, 64, 64, 128, 128, 128,
           64, 64, 1, 1, 1, 0, 0, 64, 64, 1, 0, 1, 0, ACT_NONE, stream);
  run_conv(bufA, wp_dr2a, nullptr, nullptr, bufB, nullptr, 16, 64, 64, 128, 128, 128,
           64, 64, 3, 3, 1, 1, 1, 64, 64, 1, 0, 1, 0, ACT_RELU, stream);
  run_conv(bufB, wp_dr2b, nullptr, bufA, bufC, nullptr, 16, 64, 64, 128, 128, 128,
           64, 64, 1, 1, 1, 0, 0, 64, 64, 1, 0, 1, 0, ACT_NONE, stream);
  lrelu_bf16<<<(8388608 + 255) / 256, 256, 0, stream>>>(bufC, 8388608L);

  // convT dt1 (128->64, 64x64 -> 128x128): 4 parity-class k2/s1 convs
  for (int par = 0; par < 4; ++par) {
    const int oy = par >> 1, ox = par & 1;
    run_conv(bufC, wp_dt1 + (size_t)par * 64 * 512, dt1_b, nullptr, bufA, nullptr,
             16, 64, 64, 128, 64, 64, 64, 64, 2, 2, 1,
             (oy == 0 ? 1 : 0), (ox == 0 ? 1 : 0),
             128, 128, 2, oy, 2, ox, ACT_LRELU, stream);
  }
  // convT dt2 (64->4, 128x128 -> 256x256) + relu -> d_out (NCHW f32)
  for (int par = 0; par < 4; ++par) {
    const int oy = par >> 1, ox = par & 1;
    run_conv(bufA, wp_dt2 + (size_t)par * 16 * 256, dt2_b, nullptr, nullptr, out_f,
             16, 128, 128, 64, 4, 16, 128, 128, 2, 2, 1,
             (oy == 0 ? 1 : 0), (ox == 0 ? 1 : 0),
             256, 256, 2, oy, 2, ox, ACT_RELU, stream);
  }
}
